// ReDynamicWeightsCat11_28097676050580
// MI455X (gfx1250) — matrix operand order repaired and checked by reading, not compiled, not run on hardware
//
#include <hip/hip_runtime.h>

// ---------------------------------------------------------------------------
// ReDynamicWeightsCat11 for MI455X (gfx1250, wave32, WMMA)
// GEMMs use v_wmma_f32_16x16x32_f16 with 16x64 tiles per wave (A reuse x4).
// Shared weight slabs are staged to LDS with global_load_async_to_lds_b128
// (ASYNCcnt) and read back as A fragments via DS loads.
// ---------------------------------------------------------------------------

typedef __attribute__((ext_vector_type(16))) _Float16 v16h;
typedef __attribute__((ext_vector_type(8)))  _Float16 v8h;
typedef __attribute__((ext_vector_type(8)))  float    v8f;

#define HH 96
#define WW 96
#define PP 9216        // HH*WW
#define C1 512         // outer channels
#define CM 128         // mid channels
#define NB 2           // batch
#define EPSF 1e-5f

__device__ __forceinline__ v16h mkfrag(v8h lo, v8h hi) {
  return __builtin_shufflevector(lo, hi, 0, 1, 2, 3, 4, 5, 6, 7,
                                 8, 9, 10, 11, 12, 13, 14, 15);
}

__device__ __forceinline__ v8f wmma_f16(v16h a, v16h b, v8f c) {
  return __builtin_amdgcn_wmma_f32_16x16x32_f16(
      /*neg_a=*/false, a, /*neg_b=*/false, b,
      /*c_mod=*/(short)0, c, /*reuse_a=*/false, /*reuse_b=*/false);
}

// Async copy of 16 bytes global -> LDS (per lane). ldsoff is an LDS byte
// offset (low 32 bits of the generic shared-pointer address).
__device__ __forceinline__ void async_g2l_b128(unsigned int ldsoff,
                                               const void* gsrc) {
  unsigned long long g = (unsigned long long)gsrc;
  asm volatile("global_load_async_to_lds_b128 %0, %1, off"
               :: "v"(ldsoff), "v"(g) : "memory");
}

__device__ __forceinline__ void wait_async0() {
  asm volatile("s_wait_asynccnt 0" ::: "memory");
}

// --------------------------- packing kernels -------------------------------

// w_scale1 [128,512] and w_scale3 [512,128] -> f16 row-major
__global__ void k_pack_w(const float* __restrict__ w1, const float* __restrict__ w3,
                         _Float16* __restrict__ w1h, _Float16* __restrict__ w3h) {
  int i = blockIdx.x * blockDim.x + threadIdx.x;
  if (i < CM * C1) {
    w1h[i] = (_Float16)w1[i];
    w3h[i] = (_Float16)w3[i];
  }
}

// w_cat[d] (9,128,3,3) -> wch [4][9 taps][16 rows (pad 9->16)][128] f16
__global__ void k_pack_wcat(const float* __restrict__ wa, const float* __restrict__ wb,
                            const float* __restrict__ wc, const float* __restrict__ wd,
                            _Float16* __restrict__ wch) {
  int i = blockIdx.x * blockDim.x + threadIdx.x;
  if (i >= 4 * 9 * 16 * 128) return;
  int c  = i % 128;
  int co = (i / 128) % 16;
  int t  = (i / (128 * 16)) % 9;
  int d  = i / (128 * 16 * 9);
  const float* w = (d == 0) ? wa : (d == 1) ? wb : (d == 2) ? wc : wd;
  _Float16 v = (_Float16)0.f;
  if (co < 9) v = (_Float16)w[((co * 128 + c) * 3 + t / 3) * 3 + (t % 3)];
  wch[i] = v;
}

// x [n][C1][p] f32 -> xt [n][p][C1] f16 (pixel-major for B fragments)
__global__ void k_pack_x(const float* __restrict__ x, _Float16* __restrict__ xt) {
  size_t i = (size_t)blockIdx.x * blockDim.x + threadIdx.x;
  if (i >= (size_t)NB * C1 * PP) return;
  int p = (int)(i % PP);
  int c = (int)((i / PP) % C1);
  int n = (int)(i / ((size_t)C1 * PP));
  xt[((size_t)n * PP + p) * C1 + c] = (_Float16)x[i];
}

// ------------------------------ GEMM 1 -------------------------------------
// y1[n][co(128)][p] = W1[128x512] * X[512 x p]
// one wave = 16co x 64px (4 accumulators, A fragment reused 4x), K=512
__global__ __launch_bounds__(256) void k_gemm1(const _Float16* __restrict__ w1h,
                                               const _Float16* __restrict__ xt,
                                               float* __restrict__ y1) {
  const int lane = threadIdx.x & 31;
  const int wv = threadIdx.x >> 5;          // 0..7 -> co tile
  const int co_base = wv * 16;
  const int p_base = blockIdx.x * 64;
  const int n = blockIdx.y;
  const int r = lane & 15;
  const int hl = lane >> 4;                 // 0 = lanes 0-15, 1 = lanes 16-31
  const _Float16* arow = w1h + (size_t)(co_base + r) * C1;
  const _Float16* brow0 = xt + ((size_t)n * PP + p_base + r) * C1;
  v8f acc0 = {}, acc1 = {}, acc2 = {}, acc3 = {};
  for (int k = 0; k < C1; k += 32) {
    const int kA = k + hl * 8;
    v8h a0 = *(const v8h*)(arow + kA);
    v8h a1 = *(const v8h*)(arow + kA + 16);
    v16h a = mkfrag(a0, a1);
#pragma unroll
    for (int j = 0; j < 4; ++j) {
      const _Float16* brow = brow0 + (size_t)j * 16 * C1;
      __builtin_prefetch(brow + k + 64, 0, 1);
      v8h b0 = *(const v8h*)(brow + k + hl * 8);
      v8h b1 = *(const v8h*)(brow + k + 16 + hl * 8);
      v8f* pa = (j == 0) ? &acc0 : (j == 1) ? &acc1 : (j == 2) ? &acc2 : &acc3;
      *pa = wmma_f16(a, mkfrag(b0, b1), *pa);
    }
  }
  float* yb = y1 + ((size_t)n * CM + co_base + hl * 8) * PP + (p_base + r);
#pragma unroll
  for (int j = 0; j < 4; ++j) {
    const v8f acc = (j == 0) ? acc0 : (j == 1) ? acc1 : (j == 2) ? acc2 : acc3;
#pragma unroll
    for (int i = 0; i < 8; ++i) yb[(size_t)i * PP + j * 16] = acc[i];
  }
}

// --------------------------- GroupNorm stats -------------------------------
__global__ __launch_bounds__(256) void k_stats(const float* __restrict__ y,
                                               float* __restrict__ mean,
                                               float* __restrict__ rsig,
                                               int C, int cg) {
  const int b = blockIdx.x;                 // n*32+g
  const int n = b >> 5, g = b & 31;
  const size_t base = ((size_t)n * C + (size_t)g * cg) * PP;
  const int total = cg * PP;
  float s = 0.f, s2 = 0.f;
  for (int i = threadIdx.x; i < total; i += 256) {
    float v = y[base + i];
    s += v;
    s2 += v * v;
  }
  __shared__ float sh[512];
  sh[threadIdx.x] = s;
  sh[256 + threadIdx.x] = s2;
  __syncthreads();
  for (int off = 128; off > 0; off >>= 1) {
    if (threadIdx.x < off) {
      sh[threadIdx.x] += sh[threadIdx.x + off];
      sh[256 + threadIdx.x] += sh[256 + threadIdx.x + off];
    }
    __syncthreads();
  }
  if (threadIdx.x == 0) {
    float m = sh[0] / (float)total;
    float var = sh[256] / (float)total - m * m;
    mean[b] = m;
    rsig[b] = rsqrtf(var + EPSF);
  }
}

// GN1 + ReLU -> xd f32 [n][c][p] and xdt f16 [n][p][c]
__global__ void k_apply1(const float* __restrict__ y1, const float* __restrict__ mean,
                         const float* __restrict__ rsig, const float* __restrict__ gw,
                         const float* __restrict__ gb, float* __restrict__ xd,
                         _Float16* __restrict__ xdt) {
  size_t i = (size_t)blockIdx.x * blockDim.x + threadIdx.x;
  if (i >= (size_t)NB * CM * PP) return;
  int p = (int)(i % PP);
  int c = (int)((i / PP) % CM);
  int n = (int)(i / ((size_t)CM * PP));
  int b = n * 32 + (c >> 2);                // cg = 4
  float v = (y1[i] - mean[b]) * rsig[b] * gw[c] + gb[c];
  v = v > 0.f ? v : 0.f;
  xd[i] = v;
  xdt[((size_t)n * PP + p) * CM + c] = (_Float16)v;
}

// ------------------- branch filter convs (WMMA tap-GEMMs) ------------------
// flog[d][n][k(9)][p]; per-dilation tap weights (9*16*128 f16 = 36KB) are
// async-staged to LDS once per block and shared by all 8 waves.
__global__ __launch_bounds__(256) void k_branch(const _Float16* __restrict__ wch,
                                                const _Float16* __restrict__ xdt,
                                                float* __restrict__ flog) {
  __shared__ _Float16 sW[9 * 16 * 128];     // 36864 bytes
  const int tid = threadIdx.x;
  const int lane = tid & 31;
  const int wv = tid >> 5;
  const int n = blockIdx.y;
  const int d = blockIdx.z;
  const int dil = d ? 4 * d : 1;            // 1,4,8,12
  {
    const char* gsrc = (const char*)(wch + (size_t)d * 9 * 16 * 128);
    const unsigned int lbase = (unsigned int)(unsigned long long)(&sW[0]);
#pragma unroll
    for (int it = 0; it < 9; ++it)
      async_g2l_b128(lbase + tid * 16 + it * 4096, gsrc + tid * 16 + it * 4096);
    wait_async0();
  }
  __syncthreads();

  const int p_base = blockIdx.x * 512 + wv * 64;
  const int r = lane & 15;
  const int hl = lane >> 4;
  const v16h zz = {};
  v8f acc0 = {}, acc1 = {}, acc2 = {}, acc3 = {};
  for (int t = 0; t < 9; ++t) {
    const _Float16* arow = &sW[(t * 16 + r) * CM];  // LDS reads
    // per-pixel-subtile shifted source pixels
    int ps[4];
    bool ok[4];
#pragma unroll
    for (int j = 0; j < 4; ++j) {
      const int pix = p_base + j * 16 + r;
      const int h = pix / WW + (t / 3 - 1) * dil;
      const int w = pix % WW + (t % 3 - 1) * dil;
      ok[j] = (h >= 0) && (h < HH) && (w >= 0) && (w < WW);
      ps[j] = ok[j] ? (h * WW + w) : pix;   // valid fallback address
    }
#pragma unroll
    for (int k = 0; k < CM; k += 32) {
      const int kA = k + hl * 8;
      v8h a0 = *(const v8h*)(arow + kA);
      v8h a1 = *(const v8h*)(arow + kA + 16);
      v16h a = mkfrag(a0, a1);
#pragma unroll
      for (int j = 0; j < 4; ++j) {
        const _Float16* brow = xdt + ((size_t)n * PP + ps[j]) * CM;
        v8h b0 = *(const v8h*)(brow + k + hl * 8);
        v8h b1 = *(const v8h*)(brow + k + 16 + hl * 8);
        v16h bf = mkfrag(b0, b1);
        bf = ok[j] ? bf : zz;               // zero-pad OOB (data select)
        v8f* pa = (j == 0) ? &acc0 : (j == 1) ? &acc1 : (j == 2) ? &acc2 : &acc3;
        *pa = wmma_f16(a, bf, *pa);
      }
    }
  }
#pragma unroll
  for (int j = 0; j < 4; ++j) {
    const v8f acc = (j == 0) ? acc0 : (j == 1) ? acc1 : (j == 2) ? acc2 : acc3;
    const int pix = p_base + j * 16 + r;
#pragma unroll
    for (int i = 0; i < 8; ++i) {
      const int co = hl * 8 + i;
      if (co < 9)
        flog[(((size_t)d * NB + n) * 9 + co) * PP + pix] = acc[i];
    }
  }
}

// softmax over 9 taps, fold gamma_d in
__global__ void k_softmax(float* __restrict__ flog, const float* __restrict__ g1,
                          const float* __restrict__ g2, const float* __restrict__ g3,
                          const float* __restrict__ g4) {
  int i = blockIdx.x * blockDim.x + threadIdx.x;  // 4*NB*PP
  if (i >= 4 * NB * PP) return;
  int p = i % PP;
  int n = (i / PP) % NB;
  int d = i / (PP * NB);
  float* f = flog + (((size_t)d * NB + n) * 9) * PP + p;
  float m = f[0];
#pragma unroll
  for (int k = 1; k < 9; ++k) m = fmaxf(m, f[(size_t)k * PP]);
  float e[9], s = 0.f;
#pragma unroll
  for (int k = 0; k < 9; ++k) {
    e[k] = __expf(f[(size_t)k * PP] - m);
    s += e[k];
  }
  const float gm = (d == 0) ? g1[0] : (d == 1) ? g2[0] : (d == 2) ? g3[0] : g4[0];
  const float inv = gm / s;
#pragma unroll
  for (int k = 0; k < 9; ++k) f[(size_t)k * PP] = e[k] * inv;
}

// tap mix: obt[n][p][c] = sum_{d,t} filt[d][n][t][p] * xd[n][c][shift(p,d,t)]
__global__ __launch_bounds__(128) void k_mix(const float* __restrict__ flog,
                                             const float* __restrict__ xd,
                                             _Float16* __restrict__ obt) {
  const int blk = blockIdx.x;               // n*PP + p
  const int n = blk / PP, p = blk % PP;
  const int c = threadIdx.x;
  __shared__ float fw[36];
  __shared__ int sp[36];
  if (threadIdx.x < 36) {
    const int d = threadIdx.x / 9, t = threadIdx.x % 9;
    fw[threadIdx.x] = flog[(((size_t)d * NB + n) * 9 + t) * PP + p];
    const int dil = d ? 4 * d : 1;
    const int h = p / WW + (t / 3 - 1) * dil;
    const int w = p % WW + (t % 3 - 1) * dil;
    sp[threadIdx.x] = (h >= 0 && h < HH && w >= 0 && w < WW) ? h * WW + w : -1;
  }
  __syncthreads();
  const float* xc = xd + ((size_t)n * CM + c) * PP;
  float acc = 0.f;
#pragma unroll
  for (int j = 0; j < 36; ++j) {
    const int s = sp[j];
    if (s >= 0) acc += fw[j] * xc[s];
  }
  obt[((size_t)n * PP + p) * CM + c] = (_Float16)acc;
}

// ------------------------------ GEMM 2 -------------------------------------
// y2[n][co(512)][p] = W3[512x128] * OB[128 x p]
// block stages a 128co x 128c weight slab (32KB) to LDS via async copy;
// each of 8 waves computes 16co x 64px with 4 accumulators.
__global__ __launch_bounds__(256) void k_gemm2(const _Float16* __restrict__ w3h,
                                               const _Float16* __restrict__ obt,
                                               float* __restrict__ y2) {
  __shared__ _Float16 sA[128 * 128];        // 32768 bytes
  const int tid = threadIdx.x;
  const int lane = tid & 31;
  const int wv = tid >> 5;
  const int n = blockIdx.y;
  const int co_slab = blockIdx.z * 128;
  {
    const char* gsrc = (const char*)(w3h + (size_t)co_slab * CM);
    const unsigned int lbase = (unsigned int)(unsigned long long)(&sA[0]);
#pragma unroll
    for (int it = 0; it < 8; ++it)
      async_g2l_b128(lbase + tid * 16 + it * 4096, gsrc + tid * 16 + it * 4096);
    wait_async0();
  }
  __syncthreads();

  const int co_base = wv * 16;              // within slab
  const int p_base = blockIdx.x * 64;
  const int r = lane & 15;
  const int hl = lane >> 4;
  const _Float16* arow = &sA[(co_base + r) * CM];
  const _Float16* brow0 = obt + ((size_t)n * PP + p_base + r) * CM;
  v8f acc0 = {}, acc1 = {}, acc2 = {}, acc3 = {};
#pragma unroll
  for (int k = 0; k < CM; k += 32) {
    const int kA = k + hl * 8;
    v8h a0 = *(const v8h*)(arow + kA);
    v8h a1 = *(const v8h*)(arow + kA + 16);
    v16h a = mkfrag(a0, a1);
#pragma unroll
    for (int j = 0; j < 4; ++j) {
      const _Float16* brow = brow0 + (size_t)j * 16 * CM;
      v8h b0 = *(const v8h*)(brow + k + hl * 8);
      v8h b1 = *(const v8h*)(brow + k + 16 + hl * 8);
      v8f* pa = (j == 0) ? &acc0 : (j == 1) ? &acc1 : (j == 2) ? &acc2 : &acc3;
      *pa = wmma_f16(a, mkfrag(b0, b1), *pa);
    }
  }
  float* yb = y2 + ((size_t)n * C1 + co_slab + co_base + hl * 8) * PP + (p_base + r);
#pragma unroll
  for (int j = 0; j < 4; ++j) {
    const v8f acc = (j == 0) ? acc0 : (j == 1) ? acc1 : (j == 2) ? acc2 : acc3;
#pragma unroll
    for (int i = 0; i < 8; ++i) yb[(size_t)i * PP + j * 16] = acc[i];
  }
}

// GN2 + ReLU -> d_out (f32 NCHW)
__global__ void k_apply2(const float* __restrict__ y2, const float* __restrict__ mean,
                         const float* __restrict__ rsig, const float* __restrict__ gw,
                         const float* __restrict__ gb, float* __restrict__ out) {
  size_t i = (size_t)blockIdx.x * blockDim.x + threadIdx.x;
  if (i >= (size_t)NB * C1 * PP) return;
  int c = (int)((i / PP) % C1);
  int n = (int)(i / ((size_t)C1 * PP));
  int b = n * 32 + (c >> 4);                // cg = 16
  float v = (y2[i] - mean[b]) * rsig[b] * gw[c] + gb[c];
  out[i] = v > 0.f ? v : 0.f;
}

// ------------------------------- launcher ----------------------------------

extern "C" void kernel_launch(void* const* d_in, const int* in_sizes, int n_in,
                              void* d_out, int out_size, void* d_ws, size_t ws_size,
                              hipStream_t stream) {
  const float* x    = (const float*)d_in[0];
  const float* w1   = (const float*)d_in[1];
  const float* gn1w = (const float*)d_in[2];
  const float* gn1b = (const float*)d_in[3];
  const float* wa   = (const float*)d_in[4];
  const float* wb   = (const float*)d_in[5];
  const float* wc   = (const float*)d_in[6];
  const float* wd   = (const float*)d_in[7];
  const float* g1   = (const float*)d_in[8];
  const float* g2   = (const float*)d_in[9];
  const float* g3   = (const float*)d_in[10];
  const float* g4   = (const float*)d_in[11];
  const float* w3   = (const float*)d_in[12];
  const float* gn3w = (const float*)d_in[13];
  const float* gn3b = (const float*)d_in[14];
  float* out = (float*)d_out;

  // workspace carve-up (all sizes multiples of 256 bytes; base is 256-aligned)
  char* ws = (char*)d_ws;
  _Float16* xt   = (_Float16*)ws;  ws += (size_t)NB * PP * C1 * 2;     // 18.9 MB
  _Float16* w1h  = (_Float16*)ws;  ws += (size_t)CM * C1 * 2;          // 128 KB
  _Float16* w3h  = (_Float16*)ws;  ws += (size_t)C1 * CM * 2;          // 128 KB
  _Float16* wch  = (_Float16*)ws;  ws += (size_t)4 * 9 * 16 * 128 * 2; // 144 KB
  float*    y1   = (float*)ws;     ws += (size_t)NB * CM * PP * 4;     // 9.4 MB
  float*    xd   = (float*)ws;     ws += (size_t)NB * CM * PP * 4;     // 9.4 MB
  _Float16* xdt  = (_Float16*)ws;  ws += (size_t)NB * PP * CM * 2;     // 4.7 MB
  float*    flog = (float*)ws;     ws += (size_t)4 * NB * 9 * PP * 4;  // 2.65 MB
  _Float16* obt  = (_Float16*)ws;  ws += (size_t)NB * PP * CM * 2;     // 4.7 MB
  float*    y2   = (float*)ws;     ws += (size_t)NB * C1 * PP * 4;     // 37.7 MB
  float*    mean1 = (float*)ws;    ws += 256;
  float*    rsig1 = (float*)ws;    ws += 256;
  float*    mean2 = (float*)ws;    ws += 256;
  float*    rsig2 = (float*)ws;    ws += 256;

  // 1) pack weights
  k_pack_w<<<(CM * C1 + 255) / 256, 256, 0, stream>>>(w1, w3, w1h, w3h);
  k_pack_wcat<<<(4 * 9 * 16 * 128 + 255) / 256, 256, 0, stream>>>(wa, wb, wc, wd, wch);
  // 2) pack x to pixel-major f16
  k_pack_x<<<(int)(((size_t)NB * C1 * PP + 255) / 256), 256, 0, stream>>>(x, xt);
  // 3) scale1 GEMM (WMMA, 16x64 per wave)
  k_gemm1<<<dim3(PP / 64, NB), 256, 0, stream>>>(w1h, xt, y1);
  // 4) GN1 stats + apply
  k_stats<<<NB * 32, 256, 0, stream>>>(y1, mean1, rsig1, CM, 4);
  k_apply1<<<(int)(((size_t)NB * CM * PP + 255) / 256), 256, 0, stream>>>(
      y1, mean1, rsig1, gn1w, gn1b, xd, xdt);
  // 5) branch filter convs (WMMA tap-GEMMs, LDS-staged weights)
  k_branch<<<dim3(PP / 512, NB, 4), 256, 0, stream>>>(wch, xdt, flog);
  // 6) softmax * gamma
  k_softmax<<<(4 * NB * PP + 255) / 256, 256, 0, stream>>>(flog, g1, g2, g3, g4);
  // 7) tap mix -> f16 pixel-major
  k_mix<<<NB * PP, 128, 0, stream>>>(flog, xd, obt);
  // 8) scale3 GEMM (WMMA, async LDS weight slab)
  k_gemm2<<<dim3(PP / 64, NB, 4), 256, 0, stream>>>(w3h, obt, y2);
  // 9) GN2 stats + apply -> output
  k_stats<<<NB * 32, 256, 0, stream>>>(y2, mean2, rsig2, C1, 16);
  k_apply2<<<(int)(((size_t)NB * C1 * PP + 255) / 256), 256, 0, stream>>>(
      y2, mean2, rsig2, gn3w, gn3b, out);
}
